// Vector_Quantizer_Block_5068061409692
// MI455X (gfx1250) — compile-verified, matrix operand order checked by reading, not run on hardware
//
#include <hip/hip_runtime.h>
#include <hip/hip_bf16.h>

typedef __attribute__((ext_vector_type(16))) _Float16 v16h;
typedef __attribute__((ext_vector_type(8)))  float    v8f;

#define Bn 32
#define Cn 256
#define Hn 64
#define Wn 64
#define Kn 512
#define HWn (Hn * Wn)                 // 4096
#define Ntot (Bn * HWn)               // 131072 rows
#define QELEMS (Bn * Cn * HWn)        // 33554432
#define NT 32                         // N-tiles of 16 codes (512/16)
#define KT 8                          // K-steps of 32 channels (256/32)

// ---------------------------------------------------------------------------
// Pack codebook e_i_ts[C,K] (fp32) into per-lane WMMA B-operand layout (f16).
// bpack flat index t = nt*4096 + kt*512 + lane*16 + e
// lane: N = lane&15, kk = (lane>>4)*8 ; element e -> K = e<8 ? kk+e : 16+kk+(e-8)
// value = e_i_ts[(kt*32+K)*Kn + nt*16+N]
// ---------------------------------------------------------------------------
__global__ void vq_pack_kernel(const float* __restrict__ e, _Float16* __restrict__ bpack) {
    int t = blockIdx.x * blockDim.x + threadIdx.x;
    if (t >= NT * KT * 32 * 16) return;
    int ei   = t & 15;
    int lane = (t >> 4) & 31;
    int kt   = (t >> 9) & 7;
    int nt   = t >> 12;
    int kk   = (lane >> 4) << 3;
    int N    = lane & 15;
    int K    = (ei < 8) ? (kk + ei) : (16 + kk + (ei - 8));
    int c    = kt * 32 + K;
    int k    = nt * 16 + N;
    bpack[t] = (_Float16)e[c * Kn + k];
}

// e-norms ||e_k||^2 and zero the loss accumulator (stream-ordered before main).
__global__ void vq_prep_kernel(const float* __restrict__ e, float* __restrict__ enorm,
                               float* __restrict__ accum) {
    int k = blockIdx.x * blockDim.x + threadIdx.x;
    if (k < Kn) {
        float s = 0.0f;
        for (int c = 0; c < Cn; ++c) {
            float v = e[c * Kn + k];
            s += v * v;
        }
        enorm[k] = s;
    }
    if (k == 0) accum[0] = 0.0f;
}

// ---------------------------------------------------------------------------
// Main fused kernel: 128 threads = 4 waves; each wave owns TWO 16-row blocks
// (32 rows): every codebook B-tile load feeds two independent WMMA chains,
// and B tiles are software-pipelined one k-step ahead.
// ---------------------------------------------------------------------------
__global__ __launch_bounds__(128) void vq_main_kernel(
    const float* __restrict__ x, const float* __restrict__ e,
    const _Float16* __restrict__ bpack, const float* __restrict__ enorm,
    float* __restrict__ accum, float* __restrict__ outq, float* __restrict__ outidx) {

    __shared__ int sh_idx[4][32];

    const int lane = threadIdx.x & 31;
    const int wv   = threadIdx.x >> 5;
    const int rb   = blockIdx.x * 4 + wv;       // 32-row block id
    const int n0   = rb * 32;
    const int bimg = n0 >> 12;                  // / HW
    const int hw0  = n0 & (HWn - 1);
    const int xbase = bimg * (Cn * HWn) + hw0;

    // ---- Stage two A panels: rows [n0,n0+16) and [n0+16,n0+32), f16 A layout
    const int M  = lane & 15;
    const int kk = (lane >> 4) << 3;
    v16h afull0[KT], afull1[KT];
#pragma unroll
    for (int ks = 0; ks < KT; ++ks) {
        v16h a0, a1;
#pragma unroll
        for (int ei = 0; ei < 16; ++ei) {
            int K = (ei < 8) ? (kk + ei) : (16 + kk + (ei - 8));
            int c = ks * 32 + K;
            a0[ei] = (_Float16)x[xbase + c * HWn + M];
            a1[ei] = (_Float16)x[xbase + c * HWn + 16 + M];
        }
        afull0[ks] = a0;
        afull1[ks] = a1;
    }

    // ---- Distance GEMM + running argmin (best[0..7]=block0, [8..15]=block1)
    float best[16];
    int   bestn[16];
#pragma unroll
    for (int r = 0; r < 16; ++r) { best[r] = 3.4e38f; bestn[r] = 0; }

    const int Ncol = lane & 15;
    for (int nt = 0; nt < NT; ++nt) {
        const _Float16* bp = bpack + nt * 4096 + lane * 16;
        int   n  = nt * 16 + Ncol;
        float en = enorm[n];                     // in flight during WMMA chain

        v8f acc0 = {};
        v8f acc1 = {};
        v16h bv = *(const v16h*)bp;              // prologue: tile ks=0
#pragma unroll
        for (int ks = 0; ks < KT; ++ks) {
            v16h bnext;
            if (ks + 1 < KT) {
                bnext = *(const v16h*)(bp + (ks + 1) * 512);  // prefetch next tile
            } else {
                bnext = bv;
            }
            acc0 = __builtin_amdgcn_wmma_f32_16x16x32_f16(
                false, afull0[ks], false, bv, (short)0, acc0, false, false);
            acc1 = __builtin_amdgcn_wmma_f32_16x16x32_f16(
                false, afull1[ks], false, bv, (short)0, acc1, false, false);
            bv = bnext;
        }
#pragma unroll
        for (int r = 0; r < 8; ++r) {
            float s0 = __builtin_fmaf(-2.0f, acc0[r], en);
            float s1 = __builtin_fmaf(-2.0f, acc1[r], en);
            if (s0 < best[r])     { best[r]     = s0; bestn[r]     = n; }
            if (s1 < best[8 + r]) { best[8 + r] = s1; bestn[8 + r] = n; }
        }
    }

    // ---- Cross-lane argmin within each 16-lane half (rows M=r / M=r+8) ----
#pragma unroll
    for (int m = 1; m <= 8; m <<= 1) {
#pragma unroll
        for (int r = 0; r < 16; ++r) {
            float ov = __shfl_xor(best[r],  m, 32);
            int   oi = __shfl_xor(bestn[r], m, 32);
            if (ov < best[r] || (ov == best[r] && oi < bestn[r])) {
                best[r] = ov; bestn[r] = oi;
            }
        }
    }
    if ((lane & 15) == 0) {
        int base = (lane >> 4) * 8;       // lane0 -> rows 0..7, lane16 -> rows 8..15
#pragma unroll
        for (int r = 0; r < 8; ++r) {
            sh_idx[wv][base + r]      = bestn[r];       // block 0
            sh_idx[wv][16 + base + r] = bestn[8 + r];   // block 1
        }
    }
    __syncthreads();

    // encoding indices (as float, per output dtype): 32 rows, one per lane
    outidx[n0 + lane] = (float)sh_idx[wv][lane];

    // ---- Gather codebook (fp32), write quantized, accumulate loss ----
    const int cofs  = lane >> 4;                    // 0 or 1
    const int wl    = lane & 15;
    const int ksel0 = sh_idx[wv][wl];
    const int ksel1 = sh_idx[wv][16 + wl];
    const int obase = xbase + wl;
    float lsum = 0.0f;
    for (int c = cofs; c < Cn; c += 2) {
        float q0  = e[c * Kn + ksel0];
        float q1  = e[c * Kn + ksel1];
        int  off0 = obase + c * HWn;
        int  off1 = off0 + 16;
        float x0 = x[off0];
        float x1 = x[off1];
        outq[off0] = q0;                            // quantized_st == quantized numerically
        outq[off1] = q1;
        float d0 = x0 - q0;
        float d1 = x1 - q1;
        lsum = __builtin_fmaf(d0, d0, lsum);
        lsum = __builtin_fmaf(d1, d1, lsum);
    }
#pragma unroll
    for (int m = 1; m < 32; m <<= 1) lsum += __shfl_xor(lsum, m, 32);
    if (lane == 0) atomicAdd(accum, lsum);
}

__global__ void vq_final_kernel(const float* __restrict__ accum, float* __restrict__ out) {
    if (threadIdx.x == 0 && blockIdx.x == 0) {
        float mean = accum[0] / (float)QELEMS;
        out[QELEMS]     = mean;                     // dictionary_loss
        out[QELEMS + 1] = mean;                     // commitment_loss (numerically equal)
    }
}

extern "C" void kernel_launch(void* const* d_in, const int* in_sizes, int n_in,
                              void* d_out, int out_size, void* d_ws, size_t ws_size,
                              hipStream_t stream) {
    const float* x = (const float*)d_in[0];
    const float* e = (const float*)d_in[1];
    float* out = (float*)d_out;

    // workspace layout: [0,262144) bpack f16 | [262144,264192) enorm f32 | [264192,+4) accum
    _Float16* bpack = (_Float16*)d_ws;
    float*    enorm = (float*)((char*)d_ws + 262144);
    float*    accum = (float*)((char*)d_ws + 262144 + 2048);

    vq_pack_kernel<<<(NT * KT * 32 * 16 + 255) / 256, 256, 0, stream>>>(e, bpack);
    vq_prep_kernel<<<2, 256, 0, stream>>>(e, enorm, accum);

    float* outidx = out + QELEMS + 2;
    vq_main_kernel<<<Ntot / 32 / 4, 128, 0, stream>>>(x, e, bpack, enorm, accum, out, outidx);

    vq_final_kernel<<<1, 1, 0, stream>>>(accum, out);
}
